// RnnSenderReinforce_57818849738826
// MI455X (gfx1250) — matrix operand order repaired and checked by reading, not compiled, not run on hardware
//
#include <hip/hip_runtime.h>
#include <hip/hip_fp16.h>
#include <math.h>

// ---------------------------------------------------------------------------
// RnnSenderReinforce for MI455X (gfx1250, wave32, WMMA)
// All GEMMs run on v_wmma_f32_16x16x32_f16 (f16 A/B, f32 accumulate).
// Each wave computes a 32x64 C tile (8 WMMA / K-step) with a software-
// pipelined (ping-pong register) K loop so L2 load latency hides behind the
// tensor pipe instead of s_wait_loadcnt 0x0 before every WMMA.
// Weights are converted+transposed to f16 [N,K] once per launch so both A and
// B fragments are contiguous 128-bit global loads matching the ISA layouts.
// ---------------------------------------------------------------------------

typedef __attribute__((ext_vector_type(16))) _Float16 v16h;
typedef __attribute__((ext_vector_type(8)))  _Float16 v8h;
typedef __attribute__((ext_vector_type(8)))  float    v8f;

#define B_SZ   1024
#define DIN_SZ 512
#define E_SZ   512
#define H_SZ   1024
#define V_SZ   4096
#define KCAT   (E_SZ + H_SZ)   /* 1536 */
#define MAXLEN 32
#define SEQLEN (MAXLEN + 1)    /* 33, force_eos column */

// -------------------- WMMA fragment loaders (per ISA layout) ---------------

// A: 16x32 f16, row-major A[M,K] with leading dim lda.
// lane (m = lane&15, lh = lane>>4) holds K chunks [k0+8lh .. +7] and [k0+16+8lh .. +7]
__device__ __forceinline__ v16h load_a_frag(const _Float16* __restrict__ A,
                                            int lda, int m0, int k0) {
  const int lane = threadIdx.x & 31;
  const int lh   = lane >> 4;
  const _Float16* p = A + (size_t)(m0 + (lane & 15)) * lda + k0 + 8 * lh;
  v8h lo = *(const v8h*)(p);
  v8h hi = *(const v8h*)(p + 16);
  v16h r;
#pragma unroll
  for (int i = 0; i < 8; ++i) { r[i] = lo[i]; r[i + 8] = hi[i]; }
  return r;
}

// B: 32x16 f16 (KxN). Weight stored TRANSPOSED as Bt[N,K], leading dim ldb.
// lane (n = lane&15, kh = lane>>4) holds K chunks [k0+8kh .. +7] and [k0+16+8kh .. +7]
__device__ __forceinline__ v16h load_b_frag(const _Float16* __restrict__ Bt,
                                            int ldb, int n0, int k0) {
  const int lane = threadIdx.x & 31;
  const int kh   = lane >> 4;
  const _Float16* p = Bt + (size_t)(n0 + (lane & 15)) * ldb + k0 + 8 * kh;
  v8h lo = *(const v8h*)(p);
  v8h hi = *(const v8h*)(p + 16);
  v16h r;
#pragma unroll
  for (int i = 0; i < 8; ++i) { r[i] = lo[i]; r[i + 8] = hi[i]; }
  return r;
}

// -------------------- WMMA GEMM: wave computes a 32x64 tile ----------------
// C = A[MxK] * B[KxN] + bias1 + bias2 ; optional tanh ; writes f32 Cf and/or
// f16 Ch. K loop unrolled x2 with ping-pong fragment registers: the loads for
// stage k+32 issue before the WMMAs for stage k, overlapping L2 latency.
__global__ void __launch_bounds__(256)
gemm_wmma_32x64(const _Float16* __restrict__ A, int lda,
                const _Float16* __restrict__ Bt, int ldb,
                const float* __restrict__ bias1,
                const float* __restrict__ bias2,
                float* __restrict__ Cf, int ldcf,
                _Float16* __restrict__ Ch, int ldch,
                int M, int N, int K, int do_tanh) {
  const int wave = (blockIdx.x * blockDim.x + threadIdx.x) >> 5;
  const int lane = threadIdx.x & 31;
  const int nT   = N >> 6;               // 64-wide N tiles
  const int mTile = wave / nT;
  const int nTile = wave - mTile * nT;
  const int m0 = mTile << 5;             // 32 rows per wave
  const int n0 = nTile << 6;
  if (m0 >= M) return;

  v8f acc[8];                            // acc[i*4+j]: (m0+16i, n0+16j)
#pragma unroll
  for (int q = 0; q < 8; ++q)
#pragma unroll
    for (int r = 0; r < 8; ++r) acc[q][r] = 0.0f;

  v16h a0[2], b0[4], a1[2], b1[4];

  // preload stage 0 (K is a multiple of 64 for all call sites: 512/1024/1536)
#pragma unroll
  for (int i = 0; i < 2; ++i) a0[i] = load_a_frag(A, lda, m0 + 16 * i, 0);
#pragma unroll
  for (int j = 0; j < 4; ++j) b0[j] = load_b_frag(Bt, ldb, n0 + 16 * j, 0);

  for (int k0 = 0; k0 < K; k0 += 64) {
    // issue loads for stage k0+32, then compute stage k0
#pragma unroll
    for (int i = 0; i < 2; ++i) a1[i] = load_a_frag(A, lda, m0 + 16 * i, k0 + 32);
#pragma unroll
    for (int j = 0; j < 4; ++j) b1[j] = load_b_frag(Bt, ldb, n0 + 16 * j, k0 + 32);
#pragma unroll
    for (int i = 0; i < 2; ++i)
#pragma unroll
      for (int j = 0; j < 4; ++j)
        acc[i * 4 + j] = __builtin_amdgcn_wmma_f32_16x16x32_f16(
            false, a0[i], false, b0[j], (short)0, acc[i * 4 + j], false, false);

    // issue loads for stage k0+64 (if any), then compute stage k0+32
    if (k0 + 64 < K) {
#pragma unroll
      for (int i = 0; i < 2; ++i) a0[i] = load_a_frag(A, lda, m0 + 16 * i, k0 + 64);
#pragma unroll
      for (int j = 0; j < 4; ++j) b0[j] = load_b_frag(Bt, ldb, n0 + 16 * j, k0 + 64);
    }
#pragma unroll
    for (int i = 0; i < 2; ++i)
#pragma unroll
      for (int j = 0; j < 4; ++j)
        acc[i * 4 + j] = __builtin_amdgcn_wmma_f32_16x16x32_f16(
            false, a1[i], false, b1[j], (short)0, acc[i * 4 + j], false, false);
  }

  const int lh = lane >> 4;
  const int nl = lane & 15;
#pragma unroll
  for (int i = 0; i < 2; ++i) {
#pragma unroll
    for (int j = 0; j < 4; ++j) {
      const int n = n0 + 16 * j + nl;
      float bsum = (bias1 ? bias1[n] : 0.0f) + (bias2 ? bias2[n] : 0.0f);
#pragma unroll
      for (int r = 0; r < 8; ++r) {
        const int m = m0 + 16 * i + r + 8 * lh;  // C/D layout: M = vgpr + 8*(lane>>4)
        float v = acc[i * 4 + j][r] + bsum;
        if (do_tanh) v = tanhf(v);
        if (Cf) Cf[(size_t)m * ldcf + n] = v;
        if (Ch) Ch[(size_t)m * ldch + n] = (_Float16)v;
      }
    }
  }
}

// -------------------- data prep kernels ------------------------------------

__global__ void cvt_f16(const float* __restrict__ in, _Float16* __restrict__ out, int n) {
  int i = blockIdx.x * 256 + threadIdx.x;
  if (i < n) out[i] = (_Float16)in[i];
}

// out[n*K + k] = (f16) in[k*N + n]   (weight transpose to [N,K])
__global__ void transpose_cvt(const float* __restrict__ in, _Float16* __restrict__ out,
                              int K, int N) {
  int i = blockIdx.x * 256 + threadIdx.x;
  if (i < K * N) {
    int k = i / N, n = i - k * N;
    out[(size_t)n * K + k] = (_Float16)in[i];
  }
}

// WtCat[n*1536 + k] = k<512 ? W_ih[k][n] : W_hh[k-512][n]  (both [K,H] row-major)
__global__ void build_wtcat(const float* __restrict__ Wih, const float* __restrict__ Whh,
                            _Float16* __restrict__ out) {
  int i = blockIdx.x * 256 + threadIdx.x;
  if (i < H_SZ * KCAT) {
    int n = i / KCAT, k = i - n * KCAT;
    float v = (k < E_SZ) ? Wih[(size_t)k * H_SZ + n] : Whh[(size_t)(k - E_SZ) * H_SZ + n];
    out[i] = (_Float16)v;
  }
}

// broadcast sos embedding into cols [0,512) of the [B,KCAT] input buffer
__global__ void init_sos(const float* __restrict__ sos, _Float16* __restrict__ inp) {
  int i = blockIdx.x * 256 + threadIdx.x;
  if (i < B_SZ * E_SZ) {
    int b = i >> 9, j = i & 511;
    inp[(size_t)b * KCAT + j] = (_Float16)sos[j];
  }
}

__global__ void zero_f32(float* __restrict__ p, int n) {
  int i = blockIdx.x * 256 + threadIdx.x;
  if (i < n) p[i] = 0.0f;
}

// -------------------- per-step softmax/argmax/entropy + emb gather ---------
// One block per batch row. argmax token has shifted logit 0 =>
//   lp = -logZ ; ent = logZ - sum(s*e^s)/Z.
__global__ void __launch_bounds__(256)
softmax_step(const float* __restrict__ logits,         // [B,V]
             const _Float16* __restrict__ embH,        // [V,E] f16
             _Float16* __restrict__ inp_next,          // [B,KCAT], fill cols [0,512)
             float* __restrict__ out_seq, float* __restrict__ out_lp,
             float* __restrict__ out_ent, int t) {
  __shared__ float sv[256];
  __shared__ float s2[256];
  __shared__ int   si[256];
  const int b = blockIdx.x, tid = threadIdx.x;
  const float* row = logits + (size_t)b * V_SZ;

  // max + argmax (lowest-index tie-break, like jnp.argmax)
  float mx = -3.0e38f; int mi = tid;
  for (int j = tid; j < V_SZ; j += 256) {
    float v = row[j];
    if (v > mx) { mx = v; mi = j; }
  }
  sv[tid] = mx; si[tid] = mi;
  __syncthreads();
  for (int s = 128; s > 0; s >>= 1) {
    if (tid < s) {
      float v2 = sv[tid + s]; int i2 = si[tid + s];
      if (v2 > sv[tid] || (v2 == sv[tid] && i2 < si[tid])) { sv[tid] = v2; si[tid] = i2; }
    }
    __syncthreads();
  }
  const float gmax = sv[0];
  const int   tok  = si[0];
  __syncthreads();

  // Z = sum e^s ; SZ = sum s*e^s
  float z = 0.0f, sz = 0.0f;
  for (int j = tid; j < V_SZ; j += 256) {
    float s = row[j] - gmax;
    float e = __expf(s);
    z += e; sz += s * e;
  }
  sv[tid] = z; s2[tid] = sz;
  __syncthreads();
  for (int s = 128; s > 0; s >>= 1) {
    if (tid < s) { sv[tid] += sv[tid + s]; s2[tid] += s2[tid + s]; }
    __syncthreads();
  }
  if (tid == 0) {
    float Z = sv[0];
    float logZ = __logf(Z);
    out_seq[b * SEQLEN + t] = (float)tok;
    out_lp [b * SEQLEN + t] = -logZ;
    out_ent[b * SEQLEN + t] = logZ - s2[0] / Z;
  }

  // gather next input embedding (f16) into cols [0,512)
  const _Float16* er = embH + (size_t)tok * E_SZ;
  for (int j = tid; j < E_SZ; j += 256)
    inp_next[(size_t)b * KCAT + j] = er[j];
}

// -------------------- host orchestration -----------------------------------

extern "C" void kernel_launch(void* const* d_in, const int* in_sizes, int n_in,
                              void* d_out, int out_size, void* d_ws, size_t ws_size,
                              hipStream_t stream) {
  (void)in_sizes; (void)n_in; (void)out_size; (void)ws_size;

  const float* x       = (const float*)d_in[0];
  const float* W_agent = (const float*)d_in[1];
  const float* b_agent = (const float*)d_in[2];
  const float* sos     = (const float*)d_in[3];
  const float* emb     = (const float*)d_in[4];
  const float* W_ih    = (const float*)d_in[5];
  const float* W_hh    = (const float*)d_in[6];
  const float* b_ih    = (const float*)d_in[7];
  const float* b_hh    = (const float*)d_in[8];
  const float* W_out   = (const float*)d_in[9];
  const float* b_out   = (const float*)d_in[10];

  char* ws = (char*)d_ws;
  size_t off = 0;
  auto alloc = [&](size_t bytes) -> void* {
    void* p = ws + off;
    off += (bytes + 255) & ~(size_t)255;
    return p;
  };
  _Float16* xh    = (_Float16*)alloc((size_t)B_SZ * DIN_SZ * 2);   // x in f16
  _Float16* WtA   = (_Float16*)alloc((size_t)H_SZ * DIN_SZ * 2);   // W_agent^T
  _Float16* WtCat = (_Float16*)alloc((size_t)H_SZ * KCAT * 2);     // [W_ih;W_hh]^T
  _Float16* WtOut = (_Float16*)alloc((size_t)V_SZ * H_SZ * 2);     // W_out^T
  _Float16* embH  = (_Float16*)alloc((size_t)V_SZ * E_SZ * 2);     // emb f16
  _Float16* buf0  = (_Float16*)alloc((size_t)B_SZ * KCAT * 2);     // [inp|h] ping
  _Float16* buf1  = (_Float16*)alloc((size_t)B_SZ * KCAT * 2);     // [inp|h] pong
  float*    logits= (float*)   alloc((size_t)B_SZ * V_SZ * 4);

  float* out_seq = (float*)d_out;
  float* out_lp  = out_seq + B_SZ * SEQLEN;
  float* out_ent = out_lp  + B_SZ * SEQLEN;

  // zero outputs (provides the force_eos zero column t=32)
  {
    int n = 3 * B_SZ * SEQLEN;
    zero_f32<<<(n + 255) / 256, 256, 0, stream>>>(out_seq, n);
  }

  // weight/activation precision conversion + transposes
  cvt_f16<<<(B_SZ * DIN_SZ + 255) / 256, 256, 0, stream>>>(x, xh, B_SZ * DIN_SZ);
  transpose_cvt<<<(DIN_SZ * H_SZ + 255) / 256, 256, 0, stream>>>(W_agent, WtA, DIN_SZ, H_SZ);
  build_wtcat<<<(H_SZ * KCAT + 255) / 256, 256, 0, stream>>>(W_ih, W_hh, WtCat);
  transpose_cvt<<<(H_SZ * V_SZ + 255) / 256, 256, 0, stream>>>(W_out, WtOut, H_SZ, V_SZ);
  cvt_f16<<<(V_SZ * E_SZ + 255) / 256, 256, 0, stream>>>(emb, embH, V_SZ * E_SZ);
  init_sos<<<(B_SZ * E_SZ + 255) / 256, 256, 0, stream>>>(sos, buf0);

  // h0 = x @ W_agent + b_agent  -> buf0 cols [512,1536)  (f16)
  {
    int blocks = (B_SZ / 32) * (H_SZ / 64) / 8;   // 64
    gemm_wmma_32x64<<<blocks, 256, 0, stream>>>(
        xh, DIN_SZ, WtA, DIN_SZ, b_agent, nullptr,
        nullptr, 0, buf0 + E_SZ, KCAT,
        B_SZ, H_SZ, DIN_SZ, /*tanh=*/0);
  }

  _Float16* cur = buf0;
  _Float16* nxt = buf1;
  const int rnnBlocks = (B_SZ / 32) * (H_SZ / 64) / 8;   // 64
  const int outBlocks = (B_SZ / 32) * (V_SZ / 64) / 8;   // 256

  for (int t = 0; t < MAXLEN; ++t) {
    // h_t = tanh([inp|h_{t-1}] @ [W_ih;W_hh] + b_ih + b_hh) -> nxt cols [512,1536)
    gemm_wmma_32x64<<<rnnBlocks, 256, 0, stream>>>(
        cur, KCAT, WtCat, KCAT, b_ih, b_hh,
        nullptr, 0, nxt + E_SZ, KCAT,
        B_SZ, H_SZ, KCAT, /*tanh=*/1);

    // logits = h_t @ W_out + b_out  (f32)
    gemm_wmma_32x64<<<outBlocks, 256, 0, stream>>>(
        nxt + E_SZ, KCAT, WtOut, H_SZ, b_out, nullptr,
        logits, V_SZ, nullptr, 0,
        B_SZ, V_SZ, H_SZ, /*tanh=*/0);

    // argmax / logprob / entropy + gather emb[tok] -> nxt cols [0,512)
    softmax_step<<<B_SZ, 256, 0, stream>>>(
        logits, embH, nxt, out_seq, out_lp, out_ent, t);

    _Float16* tmp = cur; cur = nxt; nxt = tmp;
  }
}
